// GRU_39333310497180
// MI455X (gfx1250) — matrix operand order repaired and checked by reading, not compiled, not run on hardware
//
#include <hip/hip_runtime.h>
#include <hip/hip_bf16.h>
#include <cstdint>
#include <cstddef>

// ---------------------------------------------------------------------------
// GRU forward for MI455X (gfx1250): bf16 WMMA GEMMs with f32 accumulation.
//   xi  = x @ W_ih^T + b_ih                (batched WMMA GEMM, 32x64/wave)
//   scan: h_t = GRU(h_{t-1}, xi_t)         (fused WMMA + gates, 1 kernel/step)
//   out = hs @ W_out^T + b_out             (batched WMMA GEMM, 32x64/wave)
//   hx_out = broadcast(h_T)
// ---------------------------------------------------------------------------

typedef __bf16 bf16;
typedef __attribute__((ext_vector_type(16))) __bf16 v16bf;
typedef __attribute__((ext_vector_type(8)))  __bf16 v8bf;
typedef __attribute__((ext_vector_type(8)))  float  v8f;

static constexpr int GN = 256;   // batch
static constexpr int GT = 256;   // time steps
static constexpr int GI = 512;   // input dim
static constexpr int GH = 512;   // hidden dim

union FragU { v16bf v; v8bf h[2]; };

#define DEVFN __device__ __forceinline__

// --- A fragment: 16x32 bf16 tile, row-major source, leading dim ldk (elems).
// Per ISA 7.12.2: lane m=lane&15, half=lane>>4; chunks at k0+half*8 and
// k0+16+half*8 (each 8 contiguous bf16 = 16 bytes).
DEVFN v16bf load_a_frag(const bf16* __restrict__ tile_base, int ldk, int k0, int lane) {
  const int m = lane & 15, half = lane >> 4;
  const bf16* p = tile_base + (size_t)m * ldk + k0 + half * 8;
  FragU u;
  u.h[0] = *reinterpret_cast<const v8bf*>(p);
  u.h[1] = *reinterpret_cast<const v8bf*>(p + 16);
  return u.v;
}

// --- B fragment: 32x16 bf16, B[k][n] = W[n][k]  (C = A * W^T).
// Lane n=lane&15, half=lane>>4 holds K chunks at k0 + half*8 and k0+16+half*8
// (two 16-byte chunks straight from W row n; no transpose needed).
DEVFN v16bf load_b_frag(const bf16* __restrict__ w_tile_base, int ldk, int k0, int lane) {
  const int n = lane & 15, half = lane >> 4;
  const bf16* p = w_tile_base + (size_t)n * ldk + k0 + half * 8;
  FragU u;
  u.h[0] = *reinterpret_cast<const v8bf*>(p);
  u.h[1] = *reinterpret_cast<const v8bf*>(p + 16);
  return u.v;
}

DEVFN v8f wmma_bf16(v16bf a, v16bf b, v8f c) {
  return __builtin_amdgcn_wmma_f32_16x16x32_bf16(
      /*neg_a=*/false, a, /*neg_b=*/false, b,
      /*c_mod=*/(short)0, c, /*reuse_a=*/false, /*reuse_b=*/false);
}

// --- A fragment built from f32 hidden state with inline reset-mask + bf16 cvt.
// Each lane's 16 values come from one batch row n = row0 + (lane&15), so the
// reset mask is a single bool per lane.
DEVFN v16bf load_h_frag(const float* __restrict__ hbase, size_t row_stride,
                        int row0, int k0, int lane,
                        const uint8_t* __restrict__ is_init, int t) {
  const int m = lane & 15, half = lane >> 4;
  const int n = row0 + m;
  const bool reset = is_init[(size_t)n * GT + t] != 0;
  const float* p = hbase + (size_t)n * row_stride + k0 + half * 8;
  FragU u;
#pragma unroll
  for (int c = 0; c < 2; ++c) {
    const float* pc = p + c * 16;
    float4 f0 = *reinterpret_cast<const float4*>(pc);
    float4 f1 = *reinterpret_cast<const float4*>(pc + 4);
    if (reset) { f0.x = f0.y = f0.z = f0.w = 0.f; f1 = f0; }
    v8bf q;
    q[0] = (bf16)f0.x; q[1] = (bf16)f0.y; q[2] = (bf16)f0.z; q[3] = (bf16)f0.w;
    q[4] = (bf16)f1.x; q[5] = (bf16)f1.y; q[6] = (bf16)f1.z; q[7] = (bf16)f1.w;
    u.h[c] = q;
  }
  return u.v;
}

// ---------------------------------------------------------------------------
// Generic GEMM: C[M,Nc] = A_bf[M,K] * W_bf[Nc,K]^T + bias.
// One wave per 32x64 output tile: 2 A fragments x 4 B fragments = 8 WMMAs per
// 12 b128 loads per k-step (vs 4 per 10 with a 16x64 tile) -> WMMA-dense.
// ---------------------------------------------------------------------------
__global__ void __launch_bounds__(256)
wmma_gemm_bias(const bf16* __restrict__ A, const bf16* __restrict__ W,
               const float* __restrict__ bias, float* __restrict__ C,
               int M, int K, int Ncols) {
  const int wid  = blockIdx.x * (blockDim.x >> 5) + (threadIdx.x >> 5);
  const int lane = threadIdx.x & 31;
  const int n64  = Ncols >> 6;
  const int mt = wid / n64, nt = wid - mt * n64;
  const int row0 = mt << 5, col0 = nt << 6;      // 32-row, 64-col tile
  if (row0 >= M) return;

  v8f acc[2][4];
#pragma unroll
  for (int h = 0; h < 2; ++h)
#pragma unroll
    for (int s = 0; s < 4; ++s) acc[h][s] = {};

  const bf16* a0 = A + (size_t)row0 * K;
  const bf16* a1 = A + (size_t)(row0 + 16) * K;
  const bf16* w0 = W + (size_t)(col0 +  0) * K;
  const bf16* w1 = W + (size_t)(col0 + 16) * K;
  const bf16* w2 = W + (size_t)(col0 + 32) * K;
  const bf16* w3 = W + (size_t)(col0 + 48) * K;

  for (int k0 = 0; k0 < K; k0 += 32) {
    v16bf fa0 = load_a_frag(a0, K, k0, lane);
    v16bf fa1 = load_a_frag(a1, K, k0, lane);
    v16bf b;
    b = load_b_frag(w0, K, k0, lane);
    acc[0][0] = wmma_bf16(fa0, b, acc[0][0]);
    acc[1][0] = wmma_bf16(fa1, b, acc[1][0]);
    b = load_b_frag(w1, K, k0, lane);
    acc[0][1] = wmma_bf16(fa0, b, acc[0][1]);
    acc[1][1] = wmma_bf16(fa1, b, acc[1][1]);
    b = load_b_frag(w2, K, k0, lane);
    acc[0][2] = wmma_bf16(fa0, b, acc[0][2]);
    acc[1][2] = wmma_bf16(fa1, b, acc[1][2]);
    b = load_b_frag(w3, K, k0, lane);
    acc[0][3] = wmma_bf16(fa0, b, acc[0][3]);
    acc[1][3] = wmma_bf16(fa1, b, acc[1][3]);
  }

  // C/D layout: element i of acc -> M = (lane>>4)*8 + i, N = lane&15.
  const int jc = lane & 15;
  const int mbl = (lane >> 4) << 3;
#pragma unroll
  for (int s = 0; s < 4; ++s) {
    const int col = col0 + s * 16 + jc;
    const float bv = bias[col];
#pragma unroll
    for (int h = 0; h < 2; ++h) {
      const int mb = row0 + h * 16 + mbl;
#pragma unroll
      for (int i = 0; i < 8; ++i)
        C[(size_t)(mb + i) * Ncols + col] = acc[h][s][i] + bv;
    }
  }
}

// ---------------------------------------------------------------------------
// Fused GRU step: mask(h) -> 3-gate WMMA GEMM -> sigmoid/tanh -> h update.
// 512 waves: wave = (m_tile in [0,16)) x (j_tile in [0,32)), 16x16 tile each,
// 48 WMMAs per wave (3 gates * K=512/32).
// hprev is addressed as hprev[n*T*H + j] (hx at t=0, hs slice t-1 otherwise).
// ---------------------------------------------------------------------------
__global__ void __launch_bounds__(256)
gru_step_kernel(const float* __restrict__ hprev,
                const float* __restrict__ xi,        // [N*T, 3H] f32
                const bf16*  __restrict__ Whh,       // [3H, H] bf16
                const float* __restrict__ b_hh,      // [3H]
                const uint8_t* __restrict__ is_init, // [N, T] bool
                float* __restrict__ hs,              // [N, T, H] f32
                int t) {
  const int wid  = blockIdx.x * (blockDim.x >> 5) + (threadIdx.x >> 5);
  const int lane = threadIdx.x & 31;
  const int mt = wid >> 5;          // N/16 = 16 tiles
  const int jt = wid & 31;          // H/16 = 32 tiles
  const int row0 = mt << 4, col0 = jt << 4;
  const size_t rs = (size_t)GT * GH;

  v8f ar = {}, az = {}, an = {};
  const bf16* wr = Whh + (size_t)(0 * GH + col0) * GH;
  const bf16* wz = Whh + (size_t)(1 * GH + col0) * GH;
  const bf16* wn = Whh + (size_t)(2 * GH + col0) * GH;

  for (int k0 = 0; k0 < GH; k0 += 32) {
    v16bf a = load_h_frag(hprev, rs, row0, k0, lane, is_init, t);
    ar = wmma_bf16(a, load_b_frag(wr, GH, k0, lane), ar);
    az = wmma_bf16(a, load_b_frag(wz, GH, k0, lane), az);
    an = wmma_bf16(a, load_b_frag(wn, GH, k0, lane), an);
  }

  const int jc = col0 + (lane & 15);
  const float bhr = b_hh[jc], bhz = b_hh[GH + jc], bhn = b_hh[2 * GH + jc];
  const int mb = row0 + ((lane >> 4) << 3);
#pragma unroll
  for (int i = 0; i < 8; ++i) {
    const int n = mb + i;
    const bool reset = is_init[(size_t)n * GT + t] != 0;
    const size_t xrow = ((size_t)n * GT + t) * (size_t)(3 * GH);
    const float r  = 1.f / (1.f + __expf(-(xi[xrow + jc] + ar[i] + bhr)));
    const float z  = 1.f / (1.f + __expf(-(xi[xrow + GH + jc] + az[i] + bhz)));
    const float ng = tanhf(xi[xrow + 2 * GH + jc] + r * (an[i] + bhn));
    const float hp = reset ? 0.f : hprev[(size_t)n * rs + jc];
    hs[((size_t)n * GT + t) * GH + jc] = (1.f - z) * ng + z * hp;
  }
}

// ---------------------------------------------------------------------------
__global__ void f32_to_bf16_kernel(const float* __restrict__ src,
                                   bf16* __restrict__ dst, size_t count) {
  size_t i = ((size_t)blockIdx.x * blockDim.x + threadIdx.x) * 2;
  if (i + 1 < count) {
    float2 f = *reinterpret_cast<const float2*>(src + i);
    dst[i]     = (bf16)f.x;
    dst[i + 1] = (bf16)f.y;
  } else if (i < count) {
    dst[i] = (bf16)src[i];
  }
}

__global__ void broadcast_hT_kernel(const float* __restrict__ hs,
                                    float* __restrict__ out2) {
  const size_t total = (size_t)GN * GT * GH;
  size_t idx = (size_t)blockIdx.x * blockDim.x + threadIdx.x;
  if (idx >= total) return;
  size_t j = idx % GH;
  size_t n = idx / ((size_t)GT * GH);
  out2[idx] = hs[((size_t)n * GT + (GT - 1)) * GH + j];
}

// ---------------------------------------------------------------------------
extern "C" void kernel_launch(void* const* d_in, const int* in_sizes, int n_in,
                              void* d_out, int out_size, void* d_ws, size_t ws_size,
                              hipStream_t stream) {
  (void)in_sizes; (void)n_in; (void)out_size; (void)ws_size;
  const float*   x       = (const float*)d_in[0];
  const float*   hx      = (const float*)d_in[1];
  const float*   W_ih    = (const float*)d_in[2];
  const float*   W_hh    = (const float*)d_in[3];
  const float*   b_ih    = (const float*)d_in[4];
  const float*   b_hh    = (const float*)d_in[5];
  const float*   W_out   = (const float*)d_in[6];
  const float*   b_out   = (const float*)d_in[7];
  const uint8_t* is_init = (const uint8_t*)d_in[8];

  float* out1 = (float*)d_out;                          // [N,T,H] projection
  float* out2 = out1 + (size_t)GN * GT * GH;            // [N,T,H] h_T broadcast

  // Workspace carve (~644 MB total).
  char* ws = (char*)d_ws;
  size_t off = 0;
  auto carve = [&](size_t bytes) -> void* {
    void* p = ws + off;
    off = (off + bytes + 255) & ~(size_t)255;
    return p;
  };
  float* xi     = (float*)carve((size_t)GN * GT * 3 * GH * sizeof(float)); // 384 MB
  float* hs     = (float*)carve((size_t)GN * GT * GH * sizeof(float));     // 128 MB
  bf16*  x_bf   = (bf16*) carve((size_t)GN * GT * GI * sizeof(bf16));      //  64 MB
  bf16*  hs_bf  = (bf16*) carve((size_t)GN * GT * GH * sizeof(bf16));      //  64 MB
  bf16*  Wih_bf = (bf16*) carve((size_t)3 * GH * GI * sizeof(bf16));
  bf16*  Whh_bf = (bf16*) carve((size_t)3 * GH * GH * sizeof(bf16));
  bf16*  Wout_bf= (bf16*) carve((size_t)GH * GH * sizeof(bf16));

  auto cvt = [&](const float* s, bf16* d, size_t cnt) {
    size_t threads = (cnt + 1) / 2;
    f32_to_bf16_kernel<<<dim3((unsigned)((threads + 255) / 256)), 256, 0, stream>>>(s, d, cnt);
  };
  cvt(x,     x_bf,   (size_t)GN * GT * GI);
  cvt(W_ih,  Wih_bf, (size_t)3 * GH * GI);
  cvt(W_hh,  Whh_bf, (size_t)3 * GH * GH);
  cvt(W_out, Wout_bf,(size_t)GH * GH);

  // xi = x @ W_ih^T + b_ih : M = N*T, K = I, Nc = 3H
  {
    const int M = GN * GT, K = GI, Nc = 3 * GH;
    const int waves = (M / 32) * (Nc / 64);           // 49152
    wmma_gemm_bias<<<dim3(waves / 8), 256, 0, stream>>>(x_bf, Wih_bf, b_ih, xi, M, K, Nc);
  }

  // Sequential recurrence: one fused kernel per timestep (graph-captured).
  for (int t = 0; t < GT; ++t) {
    const float* hprev = (t == 0) ? hx : (hs + (size_t)(t - 1) * GH);
    gru_step_kernel<<<dim3(64), 256, 0, stream>>>(hprev, xi, Whh_bf, b_hh, is_init, hs, t);
  }

  // output = hs @ W_out^T + b_out
  cvt(hs, hs_bf, (size_t)GN * GT * GH);
  {
    const int M = GN * GT, K = GH, Nc = GH;
    const int waves = (M / 32) * (Nc / 64);           // 16384
    wmma_gemm_bias<<<dim3(waves / 8), 256, 0, stream>>>(hs_bf, Wout_bf, b_out, out1, M, K, Nc);
  }

  // hx_out = broadcast(h_T)
  {
    const size_t total = (size_t)GN * GT * GH;
    broadcast_hT_kernel<<<dim3((unsigned)((total + 255) / 256)), 256, 0, stream>>>(hs, out2);
  }
}
